// PerceiverAttention_70781061038872
// MI455X (gfx1250) — matrix operand order repaired and checked by reading, not compiled, not run on hardware
//
#include <hip/hip_runtime.h>
#include <hip/hip_bf16.h>

// ---------------------------------------------------------------------------
// Types / WMMA helpers (CDNA5 / gfx1250, wave32)
// ---------------------------------------------------------------------------
typedef __attribute__((ext_vector_type(16))) __bf16 v16bf;
typedef __attribute__((ext_vector_type(8)))  float  v8f;
typedef __attribute__((__vector_size__(4 * sizeof(int)))) int i32x4;

union BF16x16 { v16bf v; unsigned int u[8]; };

__device__ inline v8f v8f_zero() {
  v8f z;
#pragma unroll
  for (int r = 0; r < 8; ++r) z[r] = 0.0f;
  return z;
}

__device__ inline v8f wmma_bf16(v16bf a, v16bf b, v8f c) {
  // D = A(16x32 bf16) * B(32x16 bf16) + C(16x16 f32)
  return __builtin_amdgcn_wmma_f32_16x16x32_bf16(
      false, a, false, b, (short)0, c, false, false);
}

__device__ inline void store_out(float* p, float v)  { *p = v; }
__device__ inline void store_out(__bf16* p, float v) { *p = (__bf16)v; }

// 16B global -> LDS copy. Prefer the gfx1250 async-to-LDS path (ASYNCcnt,
// no VGPR staging); fall back to LDS-DMA or a plain VGPR round trip.
__device__ inline void async_copy16(void* lds_dst, const void* gsrc) {
#if __has_builtin(__builtin_amdgcn_global_load_async_to_lds_b128)
  __builtin_amdgcn_global_load_async_to_lds_b128(
      (__attribute__((address_space(1))) i32x4*)gsrc,
      (__attribute__((address_space(3))) i32x4*)lds_dst, 0, 0);
#elif __has_builtin(__builtin_amdgcn_global_load_lds)
  __builtin_amdgcn_global_load_lds(
      (const __attribute__((address_space(1))) void*)gsrc,
      (__attribute__((address_space(3))) void*)lds_dst, 16, 0, 0);
#else
  *(uint4*)lds_dst = *(const uint4*)gsrc;
#endif
}

__device__ inline void async_copy_wait() {
#if __has_builtin(__builtin_amdgcn_global_load_async_to_lds_b128) && \
    __has_builtin(__builtin_amdgcn_s_wait_asynccnt)
  __builtin_amdgcn_s_wait_asynccnt(0);
#endif
}

// Problem constants
#define BATCH 8
#define NCTX  4096
#define LLAT  64
#define DMODEL 1024
#define HEADS 16
#define DHEAD 64
#define NL    4160          // NCTX + LLAT
#define MKV   (BATCH * NL)  // 33280

// ---------------------------------------------------------------------------
// fp32 -> bf16 weight convert (vectorized)
// ---------------------------------------------------------------------------
__global__ __launch_bounds__(256) void cvt_kernel(const float* __restrict__ in,
                                                  __bf16* __restrict__ out, int n4) {
  int i = blockIdx.x * blockDim.x + threadIdx.x;
  if (i >= n4) return;
  float4 v = ((const float4*)in)[i];
  union { __bf16 h[4]; uint2 u; } pk;
  pk.h[0] = (__bf16)v.x; pk.h[1] = (__bf16)v.y;
  pk.h[2] = (__bf16)v.z; pk.h[3] = (__bf16)v.w;
  ((uint2*)out)[i] = pk.u;
}

// ---------------------------------------------------------------------------
// LayerNorm over D=1024, fused fp32->bf16, writes into concatenated KV input
// (and optionally a compact copy for the Q projection).
// One block (256 thr) per row; each thread owns 4 elements.
// ---------------------------------------------------------------------------
__global__ __launch_bounds__(256)
void ln_bf16_kernel(const float* __restrict__ x, const float* __restrict__ g,
                    const float* __restrict__ bparm, __bf16* __restrict__ out0,
                    int out0_bstride, int out0_rowoff,
                    __bf16* __restrict__ out1, int rows_per_batch) {
  int rg = blockIdx.x;
  int bb = rg / rows_per_batch;
  int rr = rg - bb * rows_per_batch;
  int t = threadIdx.x;

  float4 xv = ((const float4*)(x + (size_t)rg * DMODEL))[t];
  float s  = xv.x + xv.y + xv.z + xv.w;
  float s2 = xv.x * xv.x + xv.y * xv.y + xv.z * xv.z + xv.w * xv.w;
#pragma unroll
  for (int off = 16; off >= 1; off >>= 1) {
    s  += __shfl_xor(s,  off, 32);
    s2 += __shfl_xor(s2, off, 32);
  }
  __shared__ float red[2][8];
  int wave = t >> 5, lane = t & 31;
  if (lane == 0) { red[0][wave] = s; red[1][wave] = s2; }
  __syncthreads();
  s = 0.f; s2 = 0.f;
#pragma unroll
  for (int i = 0; i < 8; ++i) { s += red[0][i]; s2 += red[1][i]; }

  float mu   = s * (1.0f / DMODEL);
  float var  = fmaxf(s2 * (1.0f / DMODEL) - mu * mu, 0.0f);
  float rstd = rsqrtf(var + 1e-5f);

  float4 gv = ((const float4*)g)[t];
  float4 bv = ((const float4*)bparm)[t];
  union { __bf16 h[4]; uint2 u; } pk;
  pk.h[0] = (__bf16)((xv.x - mu) * rstd * gv.x + bv.x);
  pk.h[1] = (__bf16)((xv.y - mu) * rstd * gv.y + bv.y);
  pk.h[2] = (__bf16)((xv.z - mu) * rstd * gv.z + bv.z);
  pk.h[3] = (__bf16)((xv.w - mu) * rstd * gv.w + bv.w);

  size_t o0 = ((size_t)bb * out0_bstride + out0_rowoff + rr) * DMODEL + t * 4;
  *(uint2*)(out0 + o0) = pk.u;
  if (out1) *(uint2*)(out1 + (size_t)rg * DMODEL + t * 4) = pk.u;
}

// ---------------------------------------------------------------------------
// Tiled WMMA GEMM: C(MxN) = A(MxK,bf16,row-major) * B(KxN,bf16,row-major)
// Block tile 128x128, 256 threads = 8 waves (4 M-waves x 2 N-waves),
// wave tile 32x64 = 2x4 WMMA 16x16 accumulators, K-step 32.
// Double-buffered LDS: tile k+1 is staged (async-to-LDS when available)
// while the 8 WMMAs of tile k execute.
// A staged [m][k]; B staged transposed [n][k] so both operands' K-pairs are
// single aligned b32 LDS reads. M,N multiples of 128; K multiple of 32.
// ---------------------------------------------------------------------------
template <typename OUT>
__global__ __launch_bounds__(256)
void gemm_bf16_wmma(const __bf16* __restrict__ A, const __bf16* __restrict__ B,
                    OUT* __restrict__ C, int M, int N, int K) {
  __shared__ __align__(16) __bf16 As[2][128 * 32];   // [m][k]
  __shared__ __align__(16) __bf16 Bs[2][128 * 34];   // [n][k] (+2 pad)

  const int tid  = threadIdx.x;
  const int wave = tid >> 5, lane = tid & 31;
  const int wm = wave >> 1, wn = wave & 1;
  const int tileM = blockIdx.y * 128, tileN = blockIdx.x * 128;

  const int lid    = lane & 15;
  const int mbase  = (lane >= 16) ? 8 : 0;   // C-layout row base
  const int akoff  = (lane >= 16) ? 8 : 0;   // A-layout K offset
  const int bkbase = (lane >= 16) ? 16 : 0;  // B-layout K base

  auto load_tiles = [&](int k0, int buf) {
    // A tile: 128 rows x 32 k, 16B per transaction (async path = no VGPRs)
    for (int i = tid; i < 512; i += 256) {
      int m = i >> 2, g = i & 3;
      async_copy16(&As[buf][m * 32 + g * 8],
                   A + (size_t)(tileM + m) * K + k0 + g * 8);
    }
    // B tile: 32 k-rows x 128 n, transposed on LDS store (manual path)
    for (int i = tid; i < 512; i += 256) {
      int kk = i >> 4, g = i & 15;
      uint4 d = *(const uint4*)(B + (size_t)(k0 + kk) * N + tileN + g * 8);
      const unsigned short* e = (const unsigned short*)&d;
      unsigned short* bsu = (unsigned short*)&Bs[buf][0];
#pragma unroll
      for (int j = 0; j < 8; ++j) bsu[(g * 8 + j) * 34 + kk] = e[j];
    }
  };

  v8f acc[2][4];
#pragma unroll
  for (int i = 0; i < 2; ++i)
#pragma unroll
    for (int j = 0; j < 4; ++j) acc[i][j] = v8f_zero();

  load_tiles(0, 0);
  async_copy_wait();
  __syncthreads();

  int buf = 0;
  for (int k0 = 0; k0 < K; k0 += 32) {
    if (k0 + 32 < K) load_tiles(k0 + 32, buf ^ 1);  // prefetch next stage

    BF16x16 afr[2], bfr[4];
#pragma unroll
    for (int mi = 0; mi < 2; ++mi) {
      int m = wm * 32 + mi * 16 + lid;
#pragma unroll
      for (int v = 0; v < 8; ++v) {
        int kb = bkbase + 2 * v;
        afr[mi].u[v] = *(const unsigned int*)&As[buf][m * 32 + kb];
      }
    }
#pragma unroll
    for (int ni = 0; ni < 4; ++ni) {
      int n = wn * 64 + ni * 16 + lid;
#pragma unroll
      for (int v = 0; v < 8; ++v) {
        int kb = bkbase + 2 * v;
        bfr[ni].u[v] = *(const unsigned int*)&Bs[buf][n * 34 + kb];
      }
    }
#pragma unroll
    for (int mi = 0; mi < 2; ++mi)
#pragma unroll
      for (int ni = 0; ni < 4; ++ni)
        acc[mi][ni] = wmma_bf16(afr[mi].v, bfr[ni].v, acc[mi][ni]);

    async_copy_wait();
    __syncthreads();
    buf ^= 1;
  }

#pragma unroll
  for (int mi = 0; mi < 2; ++mi)
#pragma unroll
    for (int ni = 0; ni < 4; ++ni)
#pragma unroll
      for (int r = 0; r < 8; ++r) {
        int row = tileM + wm * 32 + mi * 16 + mbase + r;
        int col = tileN + wn * 64 + ni * 16 + lid;
        store_out(&C[(size_t)row * N + col], acc[mi][ni][r]);
      }
}

// ---------------------------------------------------------------------------
// Flash attention: grid = b*h (128 blocks), 128 threads = 4 waves.
// Wave w owns queries [w*16, w*16+16); streams K/V in 32-key chunks via LDS
// with online softmax. QK^T and PV both via v_wmma_f32_16x16x32_bf16.
// ---------------------------------------------------------------------------
__global__ __launch_bounds__(128)
void attn_kernel(const __bf16* __restrict__ Q,   // (b,64,1024) bf16
                 const __bf16* __restrict__ KV,  // (b*4160, 2048) bf16, k|v
                 const unsigned char* __restrict__ mask,  // (b,4096) bool8
                 __bf16* __restrict__ O) {       // (b,64,1024) bf16
  __shared__ __align__(16) __bf16 Kt[32 * 64];       // [key][d]
  __shared__ __align__(16) __bf16 Vt[64 * 34];       // [d][key] (+2 pad)
  __shared__ __align__(16) __bf16 Ps[4][16 * 32];    // per-wave P scratch [m][k]

  const int tid = threadIdx.x, wave = tid >> 5, lane = tid & 31;
  const int bb = blockIdx.x >> 4, h = blockIdx.x & 15;
  const int lid    = lane & 15;
  const int mbase  = (lane >= 16) ? 8 : 0;
  const int akoff  = (lane >= 16) ? 8 : 0;
  const int bkbase = (lane >= 16) ? 16 : 0;
  const float scale = 0.125f;  // (dh^-0.25)^2, folded into the logits

  // Q fragments (16 queries x 64 dh = two 16x32 A-frags), loaded once
  BF16x16 aq[2];
  {
    const __bf16* qrow = Q + (size_t)(bb * LLAT + wave * 16 + lid) * DMODEL + h * DHEAD;
#pragma unroll
    for (int s = 0; s < 2; ++s)
#pragma unroll
      for (int v = 0; v < 8; ++v) {
        int kb = s * 32 + bkbase + 2 * v;
        aq[s].u[v] = *(const unsigned int*)(qrow + kb);
      }
  }

  v8f o[4];
#pragma unroll
  for (int ni = 0; ni < 4; ++ni) o[ni] = v8f_zero();
  float mrow[8], lrow[8];
#pragma unroll
  for (int r = 0; r < 8; ++r) { mrow[r] = -3.0e38f; lrow[r] = 0.0f; }

  const size_t kvrow0 = (size_t)bb * NL;
  const int NCHUNK = NL / 32;  // 130

  for (int c = 0; c < NCHUNK; ++c) {
    const int key0 = c * 32;
    __syncthreads();
    // Stage K (row-major [key][d], async-to-LDS when available) and
    // V (transposed [d][key], manual) chunk
    for (int i = tid; i < 256; i += 128) {
      int kk = i >> 3, g = i & 7;
      const __bf16* src = KV + (kvrow0 + key0 + kk) * 2048 + h * DHEAD;
      async_copy16(&Kt[kk * 64 + g * 8], src + g * 8);
      uint4 dv = *(const uint4*)(src + 1024 + g * 8);
      const unsigned short* e = (const unsigned short*)&dv;
      unsigned short* vtu = (unsigned short*)Vt;
#pragma unroll
      for (int j = 0; j < 8; ++j) vtu[(g * 8 + j) * 34 + kk] = e[j];
    }
    if (c + 1 < NCHUNK)  // global_prefetch_b8 of next K/V chunk
      __builtin_prefetch(KV + (kvrow0 + key0 + 32 + (tid & 31)) * 2048 + h * DHEAD, 0, 1);
    async_copy_wait();
    __syncthreads();

    // Scores: S(16x32) = Q(16x64) * Kchunk^T, two 16x16 tiles
    v8f s0 = v8f_zero(), s1 = v8f_zero();
#pragma unroll
    for (int sub = 0; sub < 2; ++sub) {
      int nkey = sub * 16 + lid;
#pragma unroll
      for (int dstep = 0; dstep < 2; ++dstep) {
        BF16x16 bk;
#pragma unroll
        for (int v = 0; v < 8; ++v) {
          int kb = dstep * 32 + bkbase + 2 * v;  // d index
          bk.u[v] = *(const unsigned int*)&Kt[nkey * 64 + kb];
        }
        if (sub == 0) s0 = wmma_bf16(aq[dstep].v, bk.v, s0);
        else          s1 = wmma_bf16(aq[dstep].v, bk.v, s1);
      }
    }

    // Mask + scale
    int key_a = key0 + lid, key_b = key0 + 16 + lid;
    bool va = (key_a >= NCTX) || (mask[bb * NCTX + key_a] != 0);
    bool vb = (key_b >= NCTX) || (mask[bb * NCTX + key_b] != 0);
    float cmax[8];
#pragma unroll
    for (int r = 0; r < 8; ++r) {
      float sa = va ? (float)s0[r] * scale : -3.0e38f;
      float sb = vb ? (float)s1[r] * scale : -3.0e38f;
      s0[r] = sa; s1[r] = sb;
      cmax[r] = fmaxf(sa, sb);
    }
#pragma unroll
    for (int off = 1; off <= 8; off <<= 1)
#pragma unroll
      for (int r = 0; r < 8; ++r)
        cmax[r] = fmaxf(cmax[r], __shfl_xor(cmax[r], off, 32));

    // Online softmax update
    float p0[8], p1[8], alpha[8], rsv[8];
#pragma unroll
    for (int r = 0; r < 8; ++r) {
      float mn = fmaxf(mrow[r], cmax[r]);
      alpha[r] = __expf(mrow[r] - mn);
      mrow[r] = mn;
      p0[r] = __expf((float)s0[r] - mn);
      p1[r] = __expf((float)s1[r] - mn);
      rsv[r] = p0[r] + p1[r];
    }
#pragma unroll
    for (int off = 1; off <= 8; off <<= 1)
#pragma unroll
      for (int r = 0; r < 8; ++r) rsv[r] += __shfl_xor(rsv[r], off, 32);
#pragma unroll
    for (int r = 0; r < 8; ++r) lrow[r] = lrow[r] * alpha[r] + rsv[r];
#pragma unroll
    for (int ni = 0; ni < 4; ++ni)
#pragma unroll
      for (int r = 0; r < 8; ++r) o[ni][r] *= alpha[r];

    // Re-layout P (C-layout -> A-layout) through per-wave LDS scratch
#pragma unroll
    for (int r = 0; r < 8; ++r) {
      Ps[wave][(mbase + r) * 32 + lid]      = (__bf16)p0[r];
      Ps[wave][(mbase + r) * 32 + 16 + lid] = (__bf16)p1[r];
    }
    asm volatile("s_wait_dscnt 0" ::: "memory");  // wave-local LDS RAW fence

    BF16x16 ap;
#pragma unroll
    for (int v = 0; v < 8; ++v) {
      int kb = bkbase + 2 * v;
      ap.u[v] = *(const unsigned int*)&Ps[wave][lid * 32 + kb];
    }
    // O(16x64) += P(16x32) * V(32x64): 4 PV WMMAs
#pragma unroll
    for (int ni = 0; ni < 4; ++ni) {
      BF16x16 bv;
      int nd = ni * 16 + lid;
#pragma unroll
      for (int v = 0; v < 8; ++v) {
        int kk = bkbase + 2 * v;  // key index
        bv.u[v] = *(const unsigned int*)&Vt[nd * 34 + kk];
      }
      o[ni] = wmma_bf16(ap.v, bv.v, o[ni]);
    }
  }

  // Epilogue: normalize and write (b, l, h*64+d) bf16
#pragma unroll
  for (int ni = 0; ni < 4; ++ni)
#pragma unroll
    for (int r = 0; r < 8; ++r) {
      int lq = wave * 16 + mbase + r;
      int dc = ni * 16 + lid;
      float val = (float)o[ni][r] / lrow[r];
      O[(size_t)(bb * LLAT + lq) * DMODEL + h * DHEAD + dc] = (__bf16)val;
    }
}

// ---------------------------------------------------------------------------
// Host launcher
// ---------------------------------------------------------------------------
extern "C" void kernel_launch(void* const* d_in, const int* in_sizes, int n_in,
                              void* d_out, int out_size, void* d_ws, size_t ws_size,
                              hipStream_t stream) {
  (void)in_sizes; (void)n_in; (void)out_size; (void)ws_size;
  const float* x    = (const float*)d_in[0];
  const float* lat  = (const float*)d_in[1];
  const unsigned char* mask = (const unsigned char*)d_in[2];  // jax bool -> 1 byte
  const float* g1   = (const float*)d_in[3];
  const float* b1   = (const float*)d_in[4];
  const float* g2   = (const float*)d_in[5];
  const float* b2   = (const float*)d_in[6];
  const float* Wq   = (const float*)d_in[7];
  const float* Wkv  = (const float*)d_in[8];
  const float* Wout = (const float*)d_in[9];
  float* out = (float*)d_out;

  char* ws = (char*)d_ws;
  size_t off = 0;
  auto alloc = [&](size_t bytes) -> void* {
    void* p = ws + off;
    off = (off + bytes + 255) & ~(size_t)255;
    return p;
  };
  __bf16* kvin  = (__bf16*)alloc((size_t)MKV * DMODEL * 2);       // [xn; ln_lat] bf16
  __bf16* qin   = (__bf16*)alloc((size_t)BATCH * LLAT * DMODEL * 2);
  __bf16* wq_b  = (__bf16*)alloc((size_t)DMODEL * DMODEL * 2);
  __bf16* wkv_b = (__bf16*)alloc((size_t)DMODEL * 2 * DMODEL * 2);
  __bf16* wo_b  = (__bf16*)alloc((size_t)DMODEL * DMODEL * 2);
  __bf16* kvout = (__bf16*)alloc((size_t)MKV * 2 * DMODEL * 2);   // k|v concat cols
  __bf16* qout  = (__bf16*)alloc((size_t)BATCH * LLAT * DMODEL * 2);
  __bf16* aout  = (__bf16*)alloc((size_t)BATCH * LLAT * DMODEL * 2);

  // Weights -> bf16
  cvt_kernel<<<(DMODEL * DMODEL / 4 + 255) / 256, 256, 0, stream>>>(Wq, wq_b, DMODEL * DMODEL / 4);
  cvt_kernel<<<(DMODEL * 2 * DMODEL / 4 + 255) / 256, 256, 0, stream>>>(Wkv, wkv_b, DMODEL * 2 * DMODEL / 4);
  cvt_kernel<<<(DMODEL * DMODEL / 4 + 255) / 256, 256, 0, stream>>>(Wout, wo_b, DMODEL * DMODEL / 4);

  // LayerNorms (fused bf16 downconvert + concat placement)
  ln_bf16_kernel<<<BATCH * NCTX, 256, 0, stream>>>(x, g1, b1, kvin, NL, 0, nullptr, NCTX);
  ln_bf16_kernel<<<BATCH * LLAT, 256, 0, stream>>>(lat, g2, b2, kvin, NL, NCTX, qin, LLAT);

  // kv projection: (33280 x 1024) @ (1024 x 2048) -> bf16
  dim3 gkv(2 * DMODEL / 128, MKV / 128);
  gemm_bf16_wmma<__bf16><<<gkv, 256, 0, stream>>>(kvin, wkv_b, kvout, MKV, 2 * DMODEL, DMODEL);

  // q projection: (512 x 1024) @ (1024 x 1024) -> bf16
  dim3 gq(DMODEL / 128, BATCH * LLAT / 128);
  gemm_bf16_wmma<__bf16><<<gq, 256, 0, stream>>>(qin, wq_b, qout, BATCH * LLAT, DMODEL, DMODEL);

  // Flash attention over 4160 keys per (b, h)
  attn_kernel<<<BATCH * HEADS, 128, 0, stream>>>(qout, kvout, mask, aout);

  // Output projection: (512 x 1024) @ (1024 x 1024) -> f32 d_out
  gemm_bf16_wmma<float><<<gq, 256, 0, stream>>>(aout, wo_b, out, BATCH * LLAT, DMODEL, DMODEL);
}
